// CvTAttention_63539746177473
// MI455X (gfx1250) — matrix operand order repaired and checked by reading, not compiled, not run on hardware
//
#include <hip/hip_runtime.h>

// ---------------------------------------------------------------------------
// CDNA5 (gfx1250) wave32 WMMA types
// ---------------------------------------------------------------------------
typedef __attribute__((ext_vector_type(16))) _Float16 v16h;
typedef __attribute__((ext_vector_type(8)))  _Float16 v8h;
typedef __attribute__((ext_vector_type(8)))  float    v8f;
typedef __attribute__((ext_vector_type(4)))  int      v4i;

#define CAT16(lo, hi) \
  __builtin_shufflevector(lo, hi, 0, 1, 2, 3, 4, 5, 6, 7, 8, 9, 10, 11, 12, 13, 14, 15)

static __device__ __forceinline__ v8f wmma_f16(v16h a, v16h b, v8f c) {
  // D(16x16 f32) = A(16x32 f16) * B(32x16 f16) + C
  return __builtin_amdgcn_wmma_f32_16x16x32_f16(false, a, false, b, (short)0, c,
                                                false, false);
}

// CDNA5 async global->LDS DMA (ASYNCcnt-tracked), guarded for portability.
// Builtin prototype (from clang diagnostics): (v4i AS1* src, v4i AS3* dst,
// imm offset, imm cpol).
#if defined(__HIP_DEVICE_COMPILE__) &&                                   \
    __has_builtin(__builtin_amdgcn_global_load_async_to_lds_b128) &&     \
    __has_builtin(__builtin_amdgcn_s_wait_asynccnt)
#define HAVE_ASYNC_LDS 1
typedef __attribute__((address_space(1))) v4i* gas1_v4i;
typedef __attribute__((address_space(3))) v4i* gas3_v4i;
#define ASYNC_CP16(dst_lds, src_glob)                                    \
  __builtin_amdgcn_global_load_async_to_lds_b128((gas1_v4i)(src_glob),   \
                                                 (gas3_v4i)(dst_lds), 0, 0)
#else
#define HAVE_ASYNC_LDS 0
#endif

#define N_B    16
#define N_T    1024
#define N_C    384
#define N_H    6
#define N_DH   64
#define ATT_SCALE 0.051031036307982884f  // 384^-0.5 (reference uses full dim)

// LDS pitches (halves); padded +8 so fragment reads spread across banks while
// keeping every b128 access 16-byte aligned (80B / 144B pitches).
#define PAD32  40
#define PAD64  72

// ---------------------------------------------------------------------------
// Fused depthwise 3x3 conv (groups=C, SAME) + BatchNorm(eval) for q,k,v.
// x: f32 [B,T,C] viewed as [B,32,32,C]; outputs f16 [B*T, C].
// ---------------------------------------------------------------------------
__global__ void dwconv_bn_kernel(
    const float* __restrict__ x,
    const float* __restrict__ dwq, const float* __restrict__ gq,
    const float* __restrict__ bq,  const float* __restrict__ mq,
    const float* __restrict__ vq,
    const float* __restrict__ dwk, const float* __restrict__ gk,
    const float* __restrict__ bk,  const float* __restrict__ mk,
    const float* __restrict__ vk,
    const float* __restrict__ dwv, const float* __restrict__ gv,
    const float* __restrict__ bv,  const float* __restrict__ mv,
    const float* __restrict__ vv,
    _Float16* __restrict__ yq, _Float16* __restrict__ yk,
    _Float16* __restrict__ yv)
{
  int idx = blockIdx.x * blockDim.x + threadIdx.x;
  if (idx >= N_B * N_T * N_C) return;
  int c = idx % N_C;
  int t = (idx / N_C) & (N_T - 1);
  int b = idx / (N_C * N_T);
  int hh = t >> 5, ww = t & 31;

  float aq = 0.f, ak = 0.f, av = 0.f;
#pragma unroll
  for (int dy = -1; dy <= 1; ++dy) {
#pragma unroll
    for (int dx = -1; dx <= 1; ++dx) {
      int y2 = hh + dy, x2 = ww + dx;
      if (y2 >= 0 && y2 < 32 && x2 >= 0 && x2 < 32) {
        float xv = x[(b * N_T + (y2 << 5) + x2) * N_C + c];
        int wi = c * 9 + (dy + 1) * 3 + (dx + 1);
        aq = fmaf(xv, dwq[wi], aq);
        ak = fmaf(xv, dwk[wi], ak);
        av = fmaf(xv, dwv[wi], av);
      }
    }
  }
  const float eps = 1e-5f;
  yq[idx] = (_Float16)((aq - mq[c]) * rsqrtf(vq[c] + eps) * gq[c] + bq[c]);
  yk[idx] = (_Float16)((ak - mk[c]) * rsqrtf(vk[c] + eps) * gk[c] + bk[c]);
  yv[idx] = (_Float16)((av - mv[c]) * rsqrtf(vv[c] + eps) * gv[c] + bv[c]);
}

// ---------------------------------------------------------------------------
// Convert the four 384x384 f32 weight matrices to f16 once.
// ---------------------------------------------------------------------------
__global__ void cvt_weights_kernel(const float* __restrict__ a,
                                   const float* __restrict__ b,
                                   const float* __restrict__ c,
                                   const float* __restrict__ d,
                                   _Float16* __restrict__ oa,
                                   _Float16* __restrict__ ob,
                                   _Float16* __restrict__ oc,
                                   _Float16* __restrict__ od)
{
  int i = blockIdx.x * blockDim.x + threadIdx.x;
  if (i >= N_C * N_C) return;
  oa[i] = (_Float16)a[i];
  ob[i] = (_Float16)b[i];
  oc[i] = (_Float16)c[i];
  od[i] = (_Float16)d[i];
}

// ---------------------------------------------------------------------------
// OUT[16384 x 384] = X[16384 x 384] @ W^T  (W stored [out=384][in=384], f16)
// mode 0: write f16 to outH.   mode 1: write f32 + bias to outF.
// 4 waves per WG, 64x64 output tile, each wave owns 16 rows.
// Async path: double-buffered LDS fed by global_load_async_to_lds_b128
// (ASYNCcnt), one barrier per K-step, no VGPR round-trip.
// Fallback: software pipeline staging the next K-tile through registers.
// ---------------------------------------------------------------------------
__global__ void gemm384_kernel(const _Float16* __restrict__ X,
                               const _Float16* __restrict__ W,
                               _Float16* __restrict__ outH,
                               float* __restrict__ outF,
                               const float* __restrict__ bias,
                               int mode)
{
  int tid  = threadIdx.x;
  int wave = tid >> 5, lane = tid & 31;
  int half = lane >> 4, mr = lane & 15;
  int rb = blockIdx.x * 64;
  int cb = blockIdx.y * 64;

  // each thread owns two fixed 8-half chunks of the 64x32 tiles
  int r0 = tid >> 2;           // rows 0..31
  int r1 = r0 + 32;            // rows 32..63
  int co = (tid & 3) * 8;      // k-offset within tile

  const _Float16* Xr0 = X + (size_t)(rb + r0) * N_C + co;
  const _Float16* Xr1 = X + (size_t)(rb + r1) * N_C + co;
  const _Float16* Wr0 = W + (size_t)(cb + r0) * N_C + co;
  const _Float16* Wr1 = W + (size_t)(cb + r1) * N_C + co;

  v8f acc[4];
#pragma unroll
  for (int t = 0; t < 4; ++t)
#pragma unroll
    for (int r = 0; r < 8; ++r) acc[t][r] = 0.f;

#if HAVE_ASYNC_LDS
  __shared__ _Float16 As[2][64][PAD32];   // [buf][m][k]
  __shared__ _Float16 Bs[2][64][PAD32];   // [buf][n][k]

  // prologue: DMA K-tile 0 straight into LDS buffer 0
  ASYNC_CP16(&As[0][r0][co], Xr0);
  ASYNC_CP16(&As[0][r1][co], Xr1);
  ASYNC_CP16(&Bs[0][r0][co], Wr0);
  ASYNC_CP16(&Bs[0][r1][co], Wr1);

  for (int kt = 0; kt < 12; ++kt) {
    int cur = kt & 1;
    // own portion of tile kt landed in LDS; barrier makes all portions
    // visible AND guarantees every wave finished reading buf cur^1.
    __builtin_amdgcn_s_wait_asynccnt(0);
    __syncthreads();

    if (kt < 11) {   // DMA tile kt+1 into the other buffer during compute
      int kn = (kt + 1) * 32;
      int nxt = cur ^ 1;
      ASYNC_CP16(&As[nxt][r0][co], Xr0 + kn);
      ASYNC_CP16(&As[nxt][r1][co], Xr1 + kn);
      ASYNC_CP16(&Bs[nxt][r0][co], Wr0 + kn);
      ASYNC_CP16(&Bs[nxt][r1][co], Wr1 + kn);
    }

    const v8h* pa0 = (const v8h*)&As[cur][wave * 16 + mr][half * 16];
    const v8h* pa1 = (const v8h*)&As[cur][wave * 16 + mr][half * 16 + 8];
    v16h a = CAT16(pa0[0], pa1[0]);

    const v8h* pb0 = (const v8h*)&Bs[cur][0 * 16 + mr][half * 16];
    const v8h* pb1 = (const v8h*)&Bs[cur][1 * 16 + mr][half * 16];
    const v8h* pb2 = (const v8h*)&Bs[cur][2 * 16 + mr][half * 16];
    const v8h* pb3 = (const v8h*)&Bs[cur][3 * 16 + mr][half * 16];
    v16h bf0 = CAT16(pb0[0], pb0[1]);
    v16h bf1 = CAT16(pb1[0], pb1[1]);
    v16h bf2 = CAT16(pb2[0], pb2[1]);
    v16h bf3 = CAT16(pb3[0], pb3[1]);
    acc[0] = wmma_f16(a, bf0, acc[0]);
    acc[1] = wmma_f16(a, bf1, acc[1]);
    acc[2] = wmma_f16(a, bf2, acc[2]);
    acc[3] = wmma_f16(a, bf3, acc[3]);
  }
#else
  __shared__ _Float16 As[64][PAD32];   // [m][k]
  __shared__ _Float16 Bs[64][PAD32];   // [n][k]

  // prologue: fetch K-tile 0 into registers
  v8h ra0 = *(const v8h*)(Xr0);
  v8h ra1 = *(const v8h*)(Xr1);
  v8h rb0 = *(const v8h*)(Wr0);
  v8h rb1 = *(const v8h*)(Wr1);

  for (int kt = 0; kt < 12; ++kt) {
    *(v8h*)&As[r0][co] = ra0;
    *(v8h*)&As[r1][co] = ra1;
    *(v8h*)&Bs[r0][co] = rb0;
    *(v8h*)&Bs[r1][co] = rb1;
    __syncthreads();

    if (kt < 11) {
      int kn = (kt + 1) * 32;
      ra0 = *(const v8h*)(Xr0 + kn);
      ra1 = *(const v8h*)(Xr1 + kn);
      rb0 = *(const v8h*)(Wr0 + kn);
      rb1 = *(const v8h*)(Wr1 + kn);
    }

    const v8h* pa0 = (const v8h*)&As[wave * 16 + mr][half * 16];
    const v8h* pa1 = (const v8h*)&As[wave * 16 + mr][half * 16 + 8];
    v16h a = CAT16(pa0[0], pa1[0]);

    const v8h* pb0 = (const v8h*)&Bs[0 * 16 + mr][half * 16];
    const v8h* pb1 = (const v8h*)&Bs[1 * 16 + mr][half * 16];
    const v8h* pb2 = (const v8h*)&Bs[2 * 16 + mr][half * 16];
    const v8h* pb3 = (const v8h*)&Bs[3 * 16 + mr][half * 16];
    v16h bf0 = CAT16(pb0[0], pb0[1]);
    v16h bf1 = CAT16(pb1[0], pb1[1]);
    v16h bf2 = CAT16(pb2[0], pb2[1]);
    v16h bf3 = CAT16(pb3[0], pb3[1]);
    acc[0] = wmma_f16(a, bf0, acc[0]);
    acc[1] = wmma_f16(a, bf1, acc[1]);
    acc[2] = wmma_f16(a, bf2, acc[2]);
    acc[3] = wmma_f16(a, bf3, acc[3]);

    __syncthreads();
  }
#endif

  // C/D layout: vgpr r -> row = r + 8*half, col = lane&15
#pragma unroll
  for (int t = 0; t < 4; ++t)
#pragma unroll
    for (int r = 0; r < 8; ++r) {
      int row = rb + wave * 16 + half * 8 + r;
      int col = cb + t * 16 + mr;
      if (mode == 0) outH[(size_t)row * N_C + col] = (_Float16)acc[t][r];
      else           outF[(size_t)row * N_C + col] = acc[t][r] + bias[col];
    }
}

// ---------------------------------------------------------------------------
// Flash attention, one (64 q-rows, b, h) tile per workgroup (4 waves).
// Q fragments persist in VGPRs; K kept [kv][d] (K-contig for QK^T), V staged
// transposed [d][kv] (K-contig for P.V); P staged per-wave C->A layout.
// K/V tiles software-pipelined through registers across kv blocks.
// ---------------------------------------------------------------------------
__global__ void flash_attn_kernel(const _Float16* __restrict__ q,
                                  const _Float16* __restrict__ k,
                                  const _Float16* __restrict__ v,
                                  const int* __restrict__ mask,
                                  _Float16* __restrict__ out)
{
  __shared__ _Float16 Ks[64][PAD64];      // [kv][d]
  __shared__ _Float16 Vt[64][PAD64];      // [d][kv]   (transposed at store)
  __shared__ _Float16 Ps[4][16][PAD64];   // per-wave P patch, [m][kv]

  int tid  = threadIdx.x;
  int wave = tid >> 5, lane = tid & 31;
  int half = lane >> 4, mr = lane & 15;
  int qb = blockIdx.x;          // 0..15 query block
  int bh = blockIdx.y;          // 0..95
  int b = bh / N_H, h = bh % N_H;
  int qrow0 = qb * 64 + wave * 16;

  // persistent Q fragments (two K-steps of 32 over head_dim 64), b128 loads
  v16h qf[2];
#pragma unroll
  for (int ks = 0; ks < 2; ++ks) {
    const v8h* pq = (const v8h*)&q[(size_t)(b * N_T + qrow0 + mr) * N_C +
                                   h * N_DH + ks * 32 + half * 16];
    qf[ks] = CAT16(pq[0], pq[1]);   // contiguous 16 halves
  }

  // per-thread K/V staging: 4 chunks of 8 halves each
  int rr = tid >> 3;            // 0..15
  int ro = (tid & 7) * 8;       // 0..56
  const _Float16* kg = k + (size_t)(b * N_T + rr) * N_C + h * N_DH + ro;
  const _Float16* vg = v + (size_t)(b * N_T + rr) * N_C + h * N_DH + ro;

  v8f o[4];
  float mstate[8], lstate[8];
#pragma unroll
  for (int t = 0; t < 4; ++t)
#pragma unroll
    for (int r = 0; r < 8; ++r) o[t][r] = 0.f;
#pragma unroll
  for (int r = 0; r < 8; ++r) { mstate[r] = -3.0e38f; lstate[r] = 0.f; }

  // prologue: fetch kv-block 0
  v8h rk[4], rv[4];
#pragma unroll
  for (int cc = 0; cc < 4; ++cc) {
    rk[cc] = *(const v8h*)(kg + (size_t)(16 * cc) * N_C);
    rv[cc] = *(const v8h*)(vg + (size_t)(16 * cc) * N_C);
  }

  for (int kb = 0; kb < 16; ++kb) {
    // stage current K (row-major b128) and V (transposed scatter) into LDS
#pragma unroll
    for (int cc = 0; cc < 4; ++cc) {
      int r = rr + 16 * cc;
      *(v8h*)&Ks[r][ro] = rk[cc];
#pragma unroll
      for (int e = 0; e < 8; ++e) Vt[ro + e][r] = rv[cc][e];
    }
    __syncthreads();

    // prefetch next kv block while this one computes
    if (kb < 15) {
#pragma unroll
      for (int cc = 0; cc < 4; ++cc) {
        size_t goff = (size_t)((kb + 1) * 64 + 16 * cc) * N_C;
        rk[cc] = *(const v8h*)(kg + goff);
        rv[cc] = *(const v8h*)(vg + goff);
      }
    }

    // S = Q @ K^T : B[k=d][n=kv] = Ks[kv][d] -> K-contiguous rows
    v8f s[4];
#pragma unroll
    for (int t = 0; t < 4; ++t)
#pragma unroll
      for (int r = 0; r < 8; ++r) s[t][r] = 0.f;
#pragma unroll
    for (int ks = 0; ks < 2; ++ks) {
      const v8h* pk0 = (const v8h*)&Ks[0 * 16 + mr][ks * 32 + half * 16];
      const v8h* pk1 = (const v8h*)&Ks[1 * 16 + mr][ks * 32 + half * 16];
      const v8h* pk2 = (const v8h*)&Ks[2 * 16 + mr][ks * 32 + half * 16];
      const v8h* pk3 = (const v8h*)&Ks[3 * 16 + mr][ks * 32 + half * 16];
      v16h b0 = CAT16(pk0[0], pk0[1]);
      v16h b1 = CAT16(pk1[0], pk1[1]);
      v16h b2 = CAT16(pk2[0], pk2[1]);
      v16h b3 = CAT16(pk3[0], pk3[1]);
      s[0] = wmma_f16(qf[ks], b0, s[0]);
      s[1] = wmma_f16(qf[ks], b1, s[1]);
      s[2] = wmma_f16(qf[ks], b2, s[2]);
      s[3] = wmma_f16(qf[ks], b3, s[3]);
    }

    // scale, mask, per-row max (rows striped: row = vgpr + 8*half)
    float rowmax[8];
#pragma unroll
    for (int r = 0; r < 8; ++r) rowmax[r] = -3.0e38f;
#pragma unroll
    for (int r = 0; r < 8; ++r) {
      const int* mrow =
          mask + (size_t)(b * N_T + qrow0 + half * 8 + r) * N_T + kb * 64 + mr;
#pragma unroll
      for (int t = 0; t < 4; ++t) {
        float val = s[t][r] * ATT_SCALE;
        if (mrow[t * 16] == 0) val = -3.0e38f;
        s[t][r] = val;
        rowmax[r] = fmaxf(rowmax[r], val);
      }
    }
#pragma unroll
    for (int r = 0; r < 8; ++r)
#pragma unroll
      for (int mk = 1; mk < 16; mk <<= 1)
        rowmax[r] = fmaxf(rowmax[r], __shfl_xor(rowmax[r], mk, 32));

    // online softmax update
#pragma unroll
    for (int r = 0; r < 8; ++r) {
      float mnew = fmaxf(mstate[r], rowmax[r]);
      float corr = __expf(mstate[r] - mnew);
      mstate[r] = mnew;
      lstate[r] *= corr;
#pragma unroll
      for (int t = 0; t < 4; ++t) o[t][r] *= corr;
      float psum = 0.f;
#pragma unroll
      for (int t = 0; t < 4; ++t) {
        float p = __expf(s[t][r] - mnew);
        s[t][r] = p;
        psum += p;
      }
#pragma unroll
      for (int mk = 1; mk < 16; mk <<= 1) psum += __shfl_xor(psum, mk, 32);
      lstate[r] += psum;
    }

    // stage P (C-layout -> row-major) in this wave's private LDS patch;
    // same-wave LDS RAW ordered via DScnt by the compiler.
#pragma unroll
    for (int t = 0; t < 4; ++t)
#pragma unroll
      for (int r = 0; r < 8; ++r)
        Ps[wave][half * 8 + r][t * 16 + mr] = (_Float16)s[t][r];

    // O += P @ V : A = P rows (two b128 runs), B = Vt rows (K-contiguous)
#pragma unroll
    for (int ks = 0; ks < 2; ++ks) {
      const v8h* pp = (const v8h*)&Ps[wave][mr][ks * 32 + half * 16];
      v16h pa = CAT16(pp[0], pp[1]);
      const v8h* pv0 = (const v8h*)&Vt[0 * 16 + mr][ks * 32 + half * 16];
      const v8h* pv1 = (const v8h*)&Vt[1 * 16 + mr][ks * 32 + half * 16];
      const v8h* pv2 = (const v8h*)&Vt[2 * 16 + mr][ks * 32 + half * 16];
      const v8h* pv3 = (const v8h*)&Vt[3 * 16 + mr][ks * 32 + half * 16];
      v16h b0 = CAT16(pv0[0], pv0[1]);
      v16h b1 = CAT16(pv1[0], pv1[1]);
      v16h b2 = CAT16(pv2[0], pv2[1]);
      v16h b3 = CAT16(pv3[0], pv3[1]);
      o[0] = wmma_f16(pa, b0, o[0]);
      o[1] = wmma_f16(pa, b1, o[1]);
      o[2] = wmma_f16(pa, b2, o[2]);
      o[3] = wmma_f16(pa, b3, o[3]);
    }

    __syncthreads();   // all waves done with Ks/Vt before next stage
  }

  // normalize and write f16 [B*T, 384]
#pragma unroll
  for (int r = 0; r < 8; ++r) {
    float inv = 1.0f / lstate[r];
    int qrow = qrow0 + half * 8 + r;
#pragma unroll
    for (int t = 0; t < 4; ++t) {
      out[(size_t)(b * N_T + qrow) * N_C + h * N_DH + t * 16 + mr] =
          (_Float16)(o[t][r] * inv);
    }
  }
}

// ---------------------------------------------------------------------------
// Launcher
// ---------------------------------------------------------------------------
extern "C" void kernel_launch(void* const* d_in, const int* in_sizes, int n_in,
                              void* d_out, int out_size, void* d_ws,
                              size_t ws_size, hipStream_t stream) {
  (void)in_sizes; (void)n_in; (void)out_size; (void)ws_size;
  const float* x    = (const float*)d_in[0];
  // d_in[1] (key) and d_in[2] (value) are unused by the reference forward.
  const int*   mask = (const int*)d_in[3];
  const float* dwq = (const float*)d_in[4];
  const float* gq  = (const float*)d_in[5];
  const float* bq  = (const float*)d_in[6];
  const float* mq  = (const float*)d_in[7];
  const float* vq  = (const float*)d_in[8];
  const float* pwq = (const float*)d_in[9];
  const float* dwk = (const float*)d_in[10];
  const float* gk  = (const float*)d_in[11];
  const float* bk  = (const float*)d_in[12];
  const float* mk  = (const float*)d_in[13];
  const float* vk  = (const float*)d_in[14];
  const float* pwk = (const float*)d_in[15];
  const float* dwv = (const float*)d_in[16];
  const float* gv  = (const float*)d_in[17];
  const float* bv  = (const float*)d_in[18];
  const float* mv  = (const float*)d_in[19];
  const float* vv  = (const float*)d_in[20];
  const float* pwv = (const float*)d_in[21];
  const float* projw = (const float*)d_in[22];
  const float* projb = (const float*)d_in[23];

  char* ws = (char*)d_ws;
  const size_t SZ  = (size_t)N_B * N_T * N_C * sizeof(_Float16);  // 12.58 MB
  const size_t WSZ = (size_t)N_C * N_C * sizeof(_Float16);
  _Float16* yq  = (_Float16*)(ws);
  _Float16* yk  = (_Float16*)(ws + 1 * SZ);
  _Float16* yv  = (_Float16*)(ws + 2 * SZ);
  _Float16* qb  = (_Float16*)(ws + 3 * SZ);
  _Float16* kb  = (_Float16*)(ws + 4 * SZ);
  _Float16* vb  = (_Float16*)(ws + 5 * SZ);
  _Float16* ao  = (_Float16*)(ws);            // reuse yq region (dead by then)
  _Float16* wpq = (_Float16*)(ws + 6 * SZ);
  _Float16* wpk = (_Float16*)(ws + 6 * SZ + 1 * WSZ);
  _Float16* wpv = (_Float16*)(ws + 6 * SZ + 2 * WSZ);
  _Float16* wpo = (_Float16*)(ws + 6 * SZ + 3 * WSZ);

  // 1) fused depthwise conv + BN -> f16
  dwconv_bn_kernel<<<(N_B * N_T * N_C) / 256, 256, 0, stream>>>(
      x, dwq, gq, bq, mq, vq, dwk, gk, bk, mk, vk, dwv, gv, bv, mv, vv,
      yq, yk, yv);

  // 2) weight conversion f32 -> f16
  cvt_weights_kernel<<<(N_C * N_C + 255) / 256, 256, 0, stream>>>(
      pwq, pwk, pwv, projw, wpq, wpk, wpv, wpo);

  // 3) pointwise 1x1 projections (WMMA GEMMs)
  dim3 gg((N_B * N_T) / 64, N_C / 64);
  gemm384_kernel<<<gg, 128, 0, stream>>>(yq, wpq, qb, nullptr, nullptr, 0);
  gemm384_kernel<<<gg, 128, 0, stream>>>(yk, wpk, kb, nullptr, nullptr, 0);
  gemm384_kernel<<<gg, 128, 0, stream>>>(yv, wpv, vb, nullptr, nullptr, 0);

  // 4) flash attention (WMMA QK^T and PV, online softmax)
  flash_attn_kernel<<<dim3(N_T / 64, N_B * N_H), 128, 0, stream>>>(
      qb, kb, vb, mask, ao);

  // 5) output projection, f32 + bias into d_out
  gemm384_kernel<<<gg, 128, 0, stream>>>(ao, wpo, nullptr, (float*)d_out,
                                         projb, 1);
}
